// TopKDictionary_88184268521507
// MI455X (gfx1250) — matrix operand order repaired and checked by reading, not compiled, not run on hardware
//
#include <hip/hip_runtime.h>
#include <hip/hip_bf16.h>

// ---------------------------------------------------------------------------
// TopK sparse autoencoder, fused:  z = x@W_enc + b_enc ; h = topk_relu(z,32) ;
// x_hat = h@W_dec + b_dec.
// v2b: 128 rows/WG (16 per wave), all waves share one B tile per step.
//      B tiles (24 KB, contiguous) DMA'd into LDS by the Tensor Data Mover,
//      double-buffered, s_wait_tensorcnt + one barrier per tile.
//      A fragments stay resident in VGPRs (24 x v16bf per wave).
//      (6-arg tensor_load_to_lds builtin for this toolchain.)
// ---------------------------------------------------------------------------

#define D_IN        768
#define NFEAT       16384
#define TOPK        32
#define BATCH       8192
#define ROWS_PER_WV 16
#define WAVES       8
#define ROWS_PER_WG (ROWS_PER_WV * WAVES)   // 128
#define KSTEPS      24                       // 768 / 32
#define FTILES      1024                     // 16384 / 16
#define TILE_HALFS  (KSTEPS * 512)           // 12288 bf16 = 24 KB per B tile
#define TILE_DWORDS (TILE_HALFS / 2)         // 6144

typedef __attribute__((ext_vector_type(16))) __bf16 v16bf;
typedef __attribute__((ext_vector_type(8)))  float  v8f;
typedef __attribute__((ext_vector_type(4)))  unsigned int v4u;
typedef __attribute__((ext_vector_type(8)))  int v8i;
typedef __attribute__((ext_vector_type(4)))  int v4i;

__device__ inline unsigned short f2bf(float f) {
    __bf16 b = (__bf16)f;
    return __builtin_bit_cast(unsigned short, b);
}

// --- prep: x [8192,768] f32 -> bf16 WMMA A-fragment-tiled layout -----------
// A 16x32 bf16 (ISA 7.12.2): lane&15 = M; half h -> k = h + (h>=8?8:0) + (lane>=16?8:0)
__global__ __launch_bounds__(256) void sae_prep_x(const float* __restrict__ x,
                                                  unsigned short* __restrict__ wsA) {
    int o    = blockIdx.x * 256 + threadIdx.x;
    int h    = o & 15;
    int lane = (o >> 4) & 31;
    int slab = o >> 9;                               // arb*24 + t
    int t    = slab % KSTEPS;
    int arb  = slab / KSTEPS;                        // global 16-row block
    int row  = arb * 16 + (lane & 15);
    int k    = t * 32 + h + ((h >= 8) ? 8 : 0) + ((lane >= 16) ? 8 : 0);
    wsA[o]   = f2bf(x[(size_t)row * D_IN + k]);
}

// --- prep: W_enc [768,16384] f32 -> bf16 WMMA B-fragment-tiled layout ------
// B 32x16 bf16: lane&15 = N; half h -> k = h + (h>=8?8:0) + (lane>=16?8:0)
// One ft tile = 24*512 halves contiguous (24 KB) => 1-D TDM transfer.
__global__ __launch_bounds__(256) void sae_prep_w(const float* __restrict__ W_enc,
                                                  unsigned short* __restrict__ wsB) {
    int o     = blockIdx.x * 256 + threadIdx.x;
    int h     = o & 15;
    int lane  = (o >> 4) & 31;
    int slab  = o >> 9;                              // ft*24 + t
    int t     = slab % KSTEPS;
    int ftile = slab / KSTEPS;
    int k     = t * 32 + h + ((h >= 8) ? 8 : 0) + ((lane >= 16) ? 8 : 0);
    int f     = ftile * 16 + (lane & 15);
    wsB[o]    = f2bf(W_enc[(size_t)k * NFEAT + f]);
}

// --- TDM: 1-D contiguous dword copy global -> LDS (ISA ch. 8 D# layout) ----
__device__ inline void tdm_load_1d(unsigned lds_off, const void* gptr, unsigned ndw) {
    unsigned long long ga = (unsigned long long)(size_t)gptr;
    v4u g0;
    g0.x = 1u;                                        // count=1 (valid), no gather
    g0.y = lds_off;                                   // lds_addr (bytes)
    g0.z = (unsigned)(ga & 0xffffffffu);              // global_addr[31:0]
    g0.w = (unsigned)((ga >> 32) & 0x01ffffffu)       // global_addr[56:32]
         | (2u << 30);                                // type=2 ("image")
    v8i g1;
    g1[0] = (int)(2u << 16);                          // wg_mask=0 | data_size=2 (4B)
    g1[1] = (int)((ndw & 0xffffu) << 16);             // tensor_dim0[15:0]
    g1[2] = (int)((ndw >> 16) | (1u << 16));          // tensor_dim0[31:16] | tensor_dim1=1
    g1[3] = (int)((ndw & 0xffffu) << 16);             // tile_dim0 = ndw
    g1[4] = 1;                                        // tile_dim1=1 | tile_dim2=0
    g1[5] = (int)ndw;                                 // tensor_dim0_stride[31:0]
    g1[6] = 0;                                        // stride hi | dim1_stride lo
    g1[7] = 0;
    v4i z4 = {0, 0, 0, 0};
    v8i z8 = {0, 0, 0, 0, 0, 0, 0, 0};
    __builtin_amdgcn_tensor_load_to_lds(g0, g1, z4, z4, z8, 0);
}

// --- fused encoder + topk + sparse decoder ---------------------------------
__global__ __launch_bounds__(256) void sae_fused(
        const unsigned short* __restrict__ wsA,
        const unsigned short* __restrict__ wsB,
        const float* __restrict__ b_enc,
        const float* __restrict__ W_dec,
        const float* __restrict__ b_dec,
        float* __restrict__ out) {

    __shared__ unsigned short bufB[2][TILE_HALFS];       // 48 KB double buffer
    __shared__ float stage[WAVES][ROWS_PER_WV][16];      //  8 KB z-tile staging
    __shared__ float lval [WAVES][ROWS_PER_WV][TOPK];    // 16 KB topk vals
    __shared__ int   lidx [WAVES][ROWS_PER_WV][TOPK];    // 16 KB topk idx

    const int tid  = threadIdx.x;
    const int wave = tid >> 5;
    const int lane = tid & 31;
    const int rb   = blockIdx.x;                         // 128-row block
    const int arb  = rb * WAVES + wave;                  // this wave's 16-row block

    // init per-wave per-row lists (owner lane = row-in-wave, lanes 0..15)
    float rmin = -1e30f; int rminp = 0;
    if (lane < ROWS_PER_WV) {
        #pragma unroll
        for (int j = 0; j < TOPK; ++j) {
            lval[wave][lane][j] = -1e30f;
            lidx[wave][lane][j] = 0;
        }
    }

    // keep this wave's A operand (16 rows x 768 k) resident in VGPRs
    const v16bf* Ab = reinterpret_cast<const v16bf*>(wsA);
    v16bf Areg[KSTEPS];
    #pragma unroll
    for (int t = 0; t < KSTEPS; ++t)
        Areg[t] = Ab[(size_t)(arb * KSTEPS + t) * 32 + lane];

    const int nsub  = lane & 15;                 // N within 16-wide feature tile
    const int rbase = (lane < 16) ? 0 : 8;       // C-layout row base per half-wave

    const unsigned lds_b0 =
        (unsigned)(size_t)(__attribute__((address_space(3))) unsigned short*)&bufB[0][0];
    const unsigned lds_b1 =
        (unsigned)(size_t)(__attribute__((address_space(3))) unsigned short*)&bufB[1][0];

    // prologue: DMA tile 0 into buffer 0
    if (wave == 0) {
        tdm_load_1d(lds_b0, wsB, TILE_DWORDS);
        __builtin_amdgcn_s_wait_tensorcnt(0);
    }
    __syncthreads();

    for (int ft = 0; ft < FTILES; ++ft) {
        const int cur = ft & 1;

        // overlap: DMA next tile into the other buffer while we compute
        if (wave == 0 && (ft + 1) < FTILES) {
            tdm_load_1d(cur ? lds_b0 : lds_b1,
                        wsB + (size_t)(ft + 1) * TILE_HALFS, TILE_DWORDS);
        }

        const int fBase = ft << 4;
        const float bias = b_enc[fBase + nsub];
        const v16bf* Bl = reinterpret_cast<const v16bf*>(&bufB[cur][0]);

        v8f c = {};
        #pragma unroll
        for (int t = 0; t < KSTEPS; ++t) {
            v16bf b = Bl[t * 32 + lane];
            c = __builtin_amdgcn_wmma_f32_16x16x32_bf16(
                    false, Areg[t], false, b, (short)0, c, false, false);
        }

        // stage z tile: VGPR r of lanes 0-15 -> row r; lanes 16-31 -> row r+8
        #pragma unroll
        for (int r = 0; r < 8; ++r)
            stage[wave][rbase + r][nsub] = c[r] + bias;

        // same-wave LDS ops are hardware in-order; block compiler reordering
        asm volatile("" ::: "memory");

        // owner lanes maintain exact running top-32 for their row
        if (lane < ROWS_PER_WV) {
            const int row = lane;
            #pragma unroll 4
            for (int n = 0; n < 16; ++n) {
                float v = stage[wave][row][n];
                if (v > rmin) {
                    lval[wave][row][rminp] = v;
                    lidx[wave][row][rminp] = fBase + n;
                    float m = lval[wave][row][0]; int mp = 0;
                    for (int j = 1; j < TOPK; ++j) {
                        float lj = lval[wave][row][j];
                        if (lj < m) { m = lj; mp = j; }
                    }
                    rmin = m; rminp = mp;
                }
            }
        }
        asm volatile("" ::: "memory");

        // wave0: next tile's DMA must be complete before anyone flips buffers
        if (wave == 0) __builtin_amdgcn_s_wait_tensorcnt(0);
        __syncthreads();
    }

    // sparse decode: x_hat = relu(topk) @ W_dec + b_dec ; 3 columns per thread
    const int c0 = tid * 3;                              // 768 = 256 * 3
    const float bd0 = b_dec[c0], bd1 = b_dec[c0 + 1], bd2 = b_dec[c0 + 2];
    for (int row = 0; row < ROWS_PER_WG; ++row) {
        const int w = row >> 4, r = row & 15;
        float a0 = bd0, a1 = bd1, a2 = bd2;
        #pragma unroll 4
        for (int j = 0; j < TOPK; ++j) {
            float coef = lval[w][r][j];
            coef = coef > 0.f ? coef : 0.f;              // relu
            const float* wd = W_dec + (size_t)lidx[w][r][j] * D_IN + c0;
            a0 += coef * wd[0];
            a1 += coef * wd[1];
            a2 += coef * wd[2];
        }
        float* op = out + (size_t)(rb * ROWS_PER_WG + row) * D_IN + c0;
        op[0] = a0; op[1] = a1; op[2] = a2;
    }
}

// ---------------------------------------------------------------------------
extern "C" void kernel_launch(void* const* d_in, const int* in_sizes, int n_in,
                              void* d_out, int out_size, void* d_ws, size_t ws_size,
                              hipStream_t stream) {
    const float* x     = (const float*)d_in[0];  // [8192, 768]
    const float* W_enc = (const float*)d_in[1];  // [768, 16384]
    const float* b_enc = (const float*)d_in[2];  // [16384]
    const float* W_dec = (const float*)d_in[3];  // [16384, 768]
    const float* b_dec = (const float*)d_in[4];  // [768]
    float* out = (float*)d_out;

    // ws layout: bf16 A-tiled x (12.6 MB) | bf16 B-tiled W_enc (25.2 MB)
    unsigned short* wsA = (unsigned short*)d_ws;
    unsigned short* wsB = (unsigned short*)((char*)d_ws + (size_t)BATCH * D_IN * 2);

    const int nA = BATCH * D_IN;    // 6,291,456 bf16 elements
    const int nB = D_IN * NFEAT;    // 12,582,912 bf16 elements

    sae_prep_x<<<nA / 256, 256, 0, stream>>>(x, wsA);
    sae_prep_w<<<nB / 256, 256, 0, stream>>>(W_enc, wsB);
    sae_fused<<<BATCH / ROWS_PER_WG, 256, 0, stream>>>(wsA, wsB, b_enc,
                                                       W_dec, b_dec, out);
}
